// STFM_79362405695532
// MI455X (gfx1250) — matrix operand order repaired and checked by reading, not compiled, not run on hardware
//
#include <hip/hip_runtime.h>
#include <hip/hip_bf16.h>

typedef __attribute__((ext_vector_type(16))) __bf16 bf16x16;
typedef __attribute__((ext_vector_type(8)))  __bf16 bf16x8;
typedef __attribute__((ext_vector_type(4)))  __bf16 bf16x4;
typedef __attribute__((ext_vector_type(8)))  float  f32x8;
typedef __attribute__((ext_vector_type(4)))  float  f32x4;

#if defined(__AMDGCN__) && __has_builtin(__builtin_amdgcn_global_load_async_to_lds_b128)
#define HAVE_ASYNC 1
typedef int i32x4 __attribute__((vector_size(16)));
#define ASYNC_LOAD_B128(gsrc, ldst) \
  __builtin_amdgcn_global_load_async_to_lds_b128((i32x4*)(gsrc), (i32x4*)(ldst), 0, 0)
#else
#define HAVE_ASYNC 0
#endif

#define LDSP  72     // row pitch (bf16) for 64-deep K tiles: 144B, multiple of 16B
#define KSTEP 64

// ---------------- sizes ----------------
#define VN      2408448      // v  [2,3,8,224,224]
#define XPN     17258496     // xp [2,768,4,2809]
#define F0N     4314624      // f0 [2,768,2809]
#define SELN    179776       // sel[2,32,2809]
#define FMATN   12945024     // fmat [2,3,2809,768]
#define KP      2848         // padded pitch for K=2841 rows
#define KMATP   13123584     // kmat/hmat [4608, 2848]
#define FC1P    8091168      // fc1_w bf16 [2841, 2848]
#define FC2P    8000032      // fc2_w bf16 [2809, 2848]
#define KPTN    12945024     // kpT [6, 2809, 768]
#define SCN     539328       // score [2,3,32,2809]
#define TEMN    49152        // tem [2,32,768]
#define TCWN    491520       // tconv_w^T [1536, 320]
#define PJWN    1769472      // proj_w [768, 2304] (already [N,K])
#define FC1N    8071281
#define FC2N    7980369
#define OUTHALF 196608

struct GemmP {
  const __bf16* A;       // direct bf16 A (modes 2,3,4)
  const __bf16* B;       // bf16 B, always stored [N,K] row-major, pitch ldb
  const float*  Af;      // fp32 source for im2col A (modes 0,1)
  const float*  scale;
  const float*  shift;
  const float*  bias;
  const float*  extra;   // ori_img (mode 0)
  float*        Cf;      // fp32 out (modes 0,1,4)
  __bf16*       Cb;      // bf16 out (modes 2,3)
  int M, N, K, lda, ldb;
};

// MODE 0: A = BN(x) rows (tconv),      epi: relu6 + ori scatter -> v
// MODE 1: A = BN(v) im2col (proj),     epi: +bias scatter -> xp
// MODE 2: A = kmat,                    epi: gelu -> bf16 hmat
// MODE 3: A = hmat,                    epi: +bias -> bf16 kpT (transposed)
// MODE 4: A = temb (batch b), B = kpT (batch z), epi: fp32 score
template<int MODE>
__global__ __launch_bounds__(128) void gemm_wmma(GemmP p) {
  __shared__ __align__(16) __bf16 sA[64 * LDSP];
  __shared__ __align__(16) __bf16 sB[64 * LDSP];

  const int tid  = threadIdx.x;
  const int lane = tid & 31;
  const int l15  = lane & 15;
  const int hi   = lane >> 4;
  const int wave = tid >> 5;
  const int wM   = (wave >> 1) * 32;
  const int wN   = (wave & 1) * 32;
  const int Mbase = blockIdx.y * 64;
  const int Nbase = blockIdx.x * 64;
  const int z = blockIdx.z;

  const __bf16* Ap = p.A;
  const __bf16* Bp = p.B;
  if constexpr (MODE == 4) {
    Ap += (long)(z / 3) * 32 * 768;
    Bp += (long)z * 2809 * 768;
  }

  f32x8 acc[2][2] = {};

  for (int kb = 0; kb < p.K; kb += KSTEP) {
    // ---- stage A (64 rows x 64 k) ----
    if constexpr (MODE == 0) {
      for (int i = tid; i < 4096; i += 128) {          // scalar; K=320, tiny GEMM
        int r = i >> 6, kk = i & 63;
        int grow = Mbase + r, gk = kb + kk;
        __bf16 bv = (__bf16)0.0f;
        if (grow < p.M) {
          int b = grow / 784; int rem = grow - b * 784;
          int t = rem / 196;  int hw = rem - t * 196;
          float v = p.Af[((b * 320 + gk) * 4 + t) * 196 + hw];
          bv = (__bf16)(v * p.scale[gk] + p.shift[gk]);
        }
        sA[r * LDSP + kk] = bv;
      }
    } else if constexpr (MODE == 1) {
      for (int i = tid; i < 1024; i += 128) {          // float4 chunks (aligned)
        int r = i >> 4, kc = (i & 15) << 2;
        int grow = Mbase + r, gk = kb + kc;            // K=2304 mult of 16
        bf16x4 o = {};
        if (grow < p.M) {
          int hw = grow % 2809; int bt = grow / 2809;
          int t = bt & 3, b = bt >> 2;
          int y = hw / 53, x = hw - y * 53;
          int kw = gk & 15, kh = (gk >> 4) & 15;
          int kt = (gk >> 8) % 3, ci = gk / 768;
          int tin = 2 * t - 1 + kt;
          if (tin >= 0 && tin < 8) {
            const float* src = p.Af + ((long)(b * 3 + ci) * 8 + tin) * 50176 +
                               (4 * y + kh) * 224 + 4 * x + kw;
            f32x4 v = *(const f32x4*)src;
            float sc = p.scale[ci], sh = p.shift[ci];
            o[0] = (__bf16)(v[0] * sc + sh); o[1] = (__bf16)(v[1] * sc + sh);
            o[2] = (__bf16)(v[2] * sc + sh); o[3] = (__bf16)(v[3] * sc + sh);
          }
        }
        *(bf16x4*)&sA[r * LDSP + kc] = o;
      }
    } else {
      for (int i = tid; i < 512; i += 128) {           // b128 chunks
        int r = i >> 3, kc = (i & 7) << 3;
        int grow = Mbase + r, gk = kb + kc;
        int lofs = r * LDSP + kc;
        if (grow < p.M && gk + 7 < p.K) {
          const __bf16* g = Ap + (long)grow * p.lda + gk;
#if HAVE_ASYNC
          ASYNC_LOAD_B128(g, &sA[lofs]);
#else
          *(bf16x8*)&sA[lofs] = *(const bf16x8*)g;
#endif
        } else {
          bf16x8 v8 = {};
          if (grow < p.M)
            for (int e = 0; e < 8; ++e)
              if (gk + e < p.K) v8[e] = Ap[(long)grow * p.lda + gk + e];
          *(bf16x8*)&sA[lofs] = v8;
        }
      }
    }
    // ---- stage B (64 n-rows x 64 k), B stored [N,K] ----
    for (int i = tid; i < 512; i += 128) {
      int n = i >> 3, kc = (i & 7) << 3;
      int gn = Nbase + n, gk = kb + kc;
      int lofs = n * LDSP + kc;
      if (gn < p.N && gk + 7 < p.K) {
        const __bf16* g = Bp + (long)gn * p.ldb + gk;
#if HAVE_ASYNC
        ASYNC_LOAD_B128(g, &sB[lofs]);
#else
        *(bf16x8*)&sB[lofs] = *(const bf16x8*)g;
#endif
      } else {
        bf16x8 v8 = {};
        if (gn < p.N)
          for (int e = 0; e < 8; ++e)
            if (gk + e < p.K) v8[e] = Bp[(long)gn * p.ldb + gk + e];
        *(bf16x8*)&sB[lofs] = v8;
      }
    }
#if HAVE_ASYNC
#if __has_builtin(__builtin_amdgcn_s_wait_asynccnt)
    __builtin_amdgcn_s_wait_asynccnt(0);
#else
    asm volatile("s_wait_asynccnt 0" ::: "memory");
#endif
#endif
    __syncthreads();

    // ---- fragments + WMMA: two 32-deep chunks per staging round ----
#pragma unroll
    for (int kh2 = 0; kh2 < 2; ++kh2) {
      bf16x16 bfrag[2];
#pragma unroll
      for (int nt = 0; nt < 2; ++nt) {
        const __bf16* br = &sB[(wN + nt * 16 + l15) * LDSP + kh2 * 32 + hi * 16];
        bf16x8 lo = *(const bf16x8*)br;
        bf16x8 hh = *(const bf16x8*)(br + 8);
        bfrag[nt] = __builtin_shufflevector(lo, hh, 0,1,2,3,4,5,6,7,8,9,10,11,12,13,14,15);
      }
#pragma unroll
      for (int mt = 0; mt < 2; ++mt) {
        const __bf16* ar = &sA[(wM + mt * 16 + l15) * LDSP + kh2 * 32];
        bf16x8 lo = *(const bf16x8*)(ar + hi * 16);
        bf16x8 hh = *(const bf16x8*)(ar + hi * 16 + 8);
        bf16x16 afrag = __builtin_shufflevector(lo, hh, 0,1,2,3,4,5,6,7,8,9,10,11,12,13,14,15);
        acc[mt][0] = __builtin_amdgcn_wmma_f32_16x16x32_bf16(false, afrag, false, bfrag[0],
                                                             (short)0, acc[mt][0], false, false);
        acc[mt][1] = __builtin_amdgcn_wmma_f32_16x16x32_bf16(false, afrag, false, bfrag[1],
                                                             (short)0, acc[mt][1], false, false);
      }
    }
    __syncthreads();
  }

  // ---- epilogue ----
#pragma unroll
  for (int mt = 0; mt < 2; ++mt)
#pragma unroll
    for (int nt = 0; nt < 2; ++nt)
#pragma unroll
      for (int r = 0; r < 8; ++r) {
        int m = Mbase + wM + mt * 16 + r + 8 * hi;
        int n = Nbase + wN + nt * 16 + l15;
        if (m >= p.M || n >= p.N) continue;
        float a = acc[mt][nt][r];
        if constexpr (MODE == 0) {
          int b = m / 784; int rem = m - b * 784;
          int t = rem / 196; int hw2 = rem - t * 196;
          int h = hw2 / 14, w = hw2 - (hw2 / 14) * 14;
          int o = n >> 9, k2 = (n >> 8) & 1, pp = (n >> 4) & 15, q = n & 15;
          int vidx = ((b * 3 + o) * 8 + (t * 2 + k2)) * 50176 + (h * 16 + pp) * 224 + (w * 16 + q);
          float val = a + p.bias[o];
          val = fminf(fmaxf(val, 0.0f), 6.0f) + p.extra[vidx];
          p.Cf[vidx] = val;
        } else if constexpr (MODE == 1) {
          int hw = m % 2809; int bt = m / 2809;
          int t = bt & 3, b = bt >> 2;
          p.Cf[((b * 768 + n) * 4 + t) * 2809 + hw] = a + p.bias[n];
        } else if constexpr (MODE == 2) {
          float h = a + p.bias[n];
          float g = 0.5f * h * (1.0f + erff(h * 0.70710678118654752f));
          p.Cb[(long)m * KP + n] = (__bf16)g;
        } else if constexpr (MODE == 3) {
          int b3t = m / 768; int c = m - b3t * 768;
          p.Cb[((long)b3t * 2809 + n) * 768 + c] = (__bf16)(a + p.bias[n]);  // transposed
        } else {
          p.Cf[(long)z * 32 * 2809 + m * 2809 + n] = a;
        }
      }
}

// ---------------- BN stats: scale = g*rsqrt(var+eps), shift = b - mean*scale ----
__global__ void bn_stats(const float* in, const float* g, const float* be,
                         float* sc, float* sh, int Bn, int S,
                         long chanStride, long batchStride) {
  int c = blockIdx.x, tid = threadIdx.x;
  __shared__ float r1[256], r2[256];
  float s = 0.f, q = 0.f;
  long tot = (long)Bn * S;
  for (long i = tid; i < tot; i += 256) {
    long b = i / S; long o = i - b * S;
    float v = in[b * batchStride + (long)c * chanStride + o];
    s += v; q += v * v;
  }
  r1[tid] = s; r2[tid] = q; __syncthreads();
  for (int st = 128; st > 0; st >>= 1) {
    if (tid < st) { r1[tid] += r1[tid + st]; r2[tid] += r2[tid + st]; }
    __syncthreads();
  }
  if (tid == 0) {
    float n = (float)tot;
    float mean = r1[0] / n;
    float var = r2[0] / n - mean * mean;
    float scv = g[c] * rsqrtf(var + 1e-5f);
    sc[c] = scv; sh[c] = be[c] - mean * scv;
  }
}

__global__ void cvt_bf16(const float* in, __bf16* out, int n) {
  int i = blockIdx.x * 256 + threadIdx.x;
  if (i < n) out[i] = (__bf16)in[i];
}

// fp32 [N,K] -> bf16 [N, KPitch]
__global__ void cvt_pad(const float* in, __bf16* out, int K, int KPitch, int total) {
  int i = blockIdx.x * 256 + threadIdx.x;
  if (i < total) { int nrow = i / K, k = i - nrow * K; out[(long)nrow * KPitch + k] = (__bf16)in[i]; }
}

__global__ void cvt_tcwT(const float* w, __bf16* out) {  // [320,1536] -> [1536,320]
  int i = blockIdx.x * 256 + threadIdx.x;
  if (i < TCWN) { int n = i / 320, k = i - n * 320; out[i] = (__bf16)w[k * 1536 + n]; }
}

// depthwise 3x3x3 pos conv, pad 1
__global__ void posconv_k(const float* xp, const float* pw, const float* pb, float* pos) {
  int idx = blockIdx.x * 256 + threadIdx.x;
  if (idx >= XPN) return;
  int hw = idx % 2809; int r = idx / 2809;
  int t = r & 3; int c = (r >> 2) % 768; int b = (r >> 2) / 768;
  int y = hw / 53, x = hw - y * 53;
  float acc = pb[c];
  for (int dt = 0; dt < 3; ++dt) {
    int tt = t + dt - 1; if (tt < 0 || tt > 3) continue;
    for (int dy = 0; dy < 3; ++dy) {
      int yy = y + dy - 1; if (yy < 0 || yy > 52) continue;
      for (int dx = 0; dx < 3; ++dx) {
        int xx = x + dx - 1; if (xx < 0 || xx > 52) continue;
        acc += xp[((b * 768 + c) * 4 + tt) * 2809 + yy * 53 + xx] *
               pw[((c * 3 + dt) * 3 + dy) * 3 + dx];
      }
    }
  }
  pos[idx] = acc;
}

__global__ void f0bn_k(const float* xp, const float* sc, const float* sh, float* f0) {
  int idx = blockIdx.x * 256 + threadIdx.x;
  if (idx >= F0N) return;
  int hw = idx % 2809; int r = idx / 2809;
  int c = r % 768; int b = r / 768;
  f0[idx] = xp[((b * 768 + c) * 4) * 2809 + hw] * sc[c] + sh[c];
}

__global__ void dwhs_k(const float* f0, const float* dww, const float* dwb, float* dsw) {
  int idx = blockIdx.x * 256 + threadIdx.x;
  if (idx >= F0N) return;
  int hw = idx % 2809; int r = idx / 2809;
  int c = r % 768; int b = r / 768;
  int y = hw / 53, x = hw - y * 53;
  float acc = dwb[c];
  for (int dy = 0; dy < 3; ++dy) {
    int yy = y + dy - 1; if (yy < 0 || yy > 52) continue;
    for (int dx = 0; dx < 3; ++dx) {
      int xx = x + dx - 1; if (xx < 0 || xx > 52) continue;
      acc += f0[(b * 768 + c) * 2809 + yy * 53 + xx] * dww[(c * 3 + dy) * 3 + dx];
    }
  }
  dsw[idx] = acc * fminf(fmaxf(acc + 3.0f, 0.0f), 6.0f) * (1.0f / 6.0f);
}

__global__ void sel_k(const float* dsw, const float* pww, const float* pwb, float* sel) {
  int idx = blockIdx.x * 256 + threadIdx.x;
  if (idx >= SELN) return;
  int hw = idx % 2809; int r = idx / 2809;
  int n = r & 31; int b = r >> 5;
  float acc = pwb[n];
  for (int c = 0; c < 768; ++c)
    acc += dsw[(b * 768 + c) * 2809 + hw] * pww[n * 768 + c];
  sel[idx] = acc;
}

// tem gather (ties preserved): one block per (b,n)
__global__ void temg_k(const float* sel, const float* f0, const float* pos,
                       float* tem, __bf16* temb, float* out) {
  int b = blockIdx.x >> 5, n = blockIdx.x & 31;
  int tid = threadIdx.x;
  __shared__ float red[256]; __shared__ float mvs;
  const float* srow = sel + (b * 32 + n) * 2809;
  float lm = -3.4e38f;
  for (int hw = tid; hw < 2809; hw += 256) lm = fmaxf(lm, srow[hw]);
  red[tid] = lm; __syncthreads();
  for (int st = 128; st > 0; st >>= 1) { if (tid < st) red[tid] = fmaxf(red[tid], red[tid + st]); __syncthreads(); }
  if (tid == 0) mvs = red[0];
  __syncthreads();
  float maxv = mvs;
  for (int j = 0; j < 3; ++j) {
    int c = tid + 256 * j;
    const float* fr = f0 + (b * 768 + c) * 2809;
    const float* pr = pos + ((b * 768 + c) * 4) * 2809;
    float af = 0.f, ap = 0.f;
    for (int hw = 0; hw < 2809; ++hw)
      if (srow[hw] == maxv) { af += fr[hw]; ap += pr[hw]; }
    int ti = (b * 32 + n) * 768 + c;
    tem[ti] = af; temb[ti] = (__bf16)af;
    int oi = ((b * 4 + 0) * 32 + n) * 768 + c;
    out[oi] = af; out[OUTHALF + oi] = ap;
  }
}

// build fmat/pmat ([b,t',hw,c], bn_c applied) and kmat hw-columns (pitch KP)
__global__ void buildf_k(const float* xp, const float* pos, const float* sc, const float* sh,
                         float* fmat, float* pmat, __bf16* kmat) {
  int idx = blockIdx.x * 256 + threadIdx.x;
  if (idx >= FMATN) return;
  int hw = idx % 2809; int r = idx / 2809;
  int c = r % 768; int r2 = r / 768;
  int t1 = r2 % 3; int b = r2 / 3;
  int xpi = ((b * 768 + c) * 4 + (t1 + 1)) * 2809 + hw;
  float fv = xp[xpi] * sc[c] + sh[c];
  int b3t = b * 3 + t1;
  fmat[((long)b3t * 2809 + hw) * 768 + c] = fv;
  pmat[((long)b3t * 2809 + hw) * 768 + c] = pos[xpi];
  kmat[((long)b3t * 768 + c) * KP + 32 + hw] = (__bf16)fv;
}

__global__ void buildq_k(const float* tem, __bf16* kmat) {
  int idx = blockIdx.x * 256 + threadIdx.x;
  if (idx >= 147456) return;
  int i = idx & 31; int r = idx >> 5;
  int c = r % 768; int r2 = r / 768;
  int t1 = r2 % 3; int b = r2 / 3;
  kmat[((long)(b * 3 + t1) * 768 + c) * KP + i] = (__bf16)tem[(b * 32 + i) * 768 + c];
}

// final gather: one block per (b,t',n)
__global__ void fing_k(const float* score, const float* fmat, const float* pmat, float* out) {
  int zi = blockIdx.x;
  int n = zi & 31; int t1 = (zi >> 5) % 3; int b = zi / 96;
  int b3t = b * 3 + t1;
  int tid = threadIdx.x;
  __shared__ float red[256]; __shared__ float mvs;
  const float* srow = score + ((long)b3t * 32 + n) * 2809;
  float lm = -3.4e38f;
  for (int hw = tid; hw < 2809; hw += 256) lm = fmaxf(lm, srow[hw]);
  red[tid] = lm; __syncthreads();
  for (int st = 128; st > 0; st >>= 1) { if (tid < st) red[tid] = fmaxf(red[tid], red[tid + st]); __syncthreads(); }
  if (tid == 0) mvs = red[0];
  __syncthreads();
  float maxv = mvs;
  for (int j = 0; j < 3; ++j) {
    int c = tid + 256 * j;
    const float* fb = fmat + (long)b3t * 2809 * 768 + c;
    const float* pb = pmat + (long)b3t * 2809 * 768 + c;
    float af = 0.f, ap = 0.f;
    for (int hw = 0; hw < 2809; ++hw)
      if (srow[hw] == maxv) { af += fb[(long)hw * 768]; ap += pb[(long)hw * 768]; }
    int oi = ((b * 4 + t1 + 1) * 32 + n) * 768 + c;
    out[oi] = af; out[OUTHALF + oi] = ap;
  }
}

// ---------------------------------------------------------------------------
extern "C" void kernel_launch(void* const* d_in, const int* in_sizes, int n_in,
                              void* d_out, int out_size, void* d_ws, size_t ws_size,
                              hipStream_t stream) {
  (void)in_sizes; (void)n_in; (void)out_size; (void)ws_size;
  const float* x      = (const float*)d_in[0];
  const float* ori    = (const float*)d_in[1];
  const float* bnag   = (const float*)d_in[2];
  const float* bnab   = (const float*)d_in[3];
  const float* tconvw = (const float*)d_in[4];
  const float* tconvb = (const float*)d_in[5];
  const float* bnpg   = (const float*)d_in[6];
  const float* bnpb   = (const float*)d_in[7];
  const float* projw  = (const float*)d_in[8];
  const float* projb  = (const float*)d_in[9];
  const float* posw   = (const float*)d_in[10];
  const float* posb   = (const float*)d_in[11];
  const float* bntg   = (const float*)d_in[12];
  const float* bntb   = (const float*)d_in[13];
  const float* dww    = (const float*)d_in[14];
  const float* dwb    = (const float*)d_in[15];
  const float* pww    = (const float*)d_in[16];
  const float* pwb    = (const float*)d_in[17];
  const float* bncg   = (const float*)d_in[18];
  const float* bncb   = (const float*)d_in[19];
  const float* fc1w   = (const float*)d_in[20];
  const float* fc1b   = (const float*)d_in[21];
  const float* fc2w   = (const float*)d_in[22];
  const float* fc2b   = (const float*)d_in[23];
  float* out = (float*)d_out;

  size_t off = 0;
  auto alloc = [&](size_t bytes) -> void* {
    off = (off + 255) & ~size_t(255);
    void* p = (char*)d_ws + off;
    off += bytes;
    return p;
  };
  float*  scaleA = (float*)alloc(320 * 4);
  float*  shiftA = (float*)alloc(320 * 4);
  float*  scaleP = (float*)alloc(8 * 4);
  float*  shiftP = (float*)alloc(8 * 4);
  float*  scaleT = (float*)alloc(768 * 4);
  float*  shiftT = (float*)alloc(768 * 4);
  float*  scaleC = (float*)alloc(768 * 4);
  float*  shiftC = (float*)alloc(768 * 4);
  __bf16* tcwbf  = (__bf16*)alloc((size_t)TCWN * 2);   // [1536,320]
  float*  vbuf   = (float*)alloc((size_t)VN * 4);
  __bf16* pjwbf  = (__bf16*)alloc((size_t)PJWN * 2);   // [768,2304]
  float*  xp     = (float*)alloc((size_t)XPN * 4);
  float*  pos    = (float*)alloc((size_t)XPN * 4);
  float*  f0bn   = (float*)alloc((size_t)F0N * 4);
  float*  dsw    = (float*)alloc((size_t)F0N * 4);
  float*  sel    = (float*)alloc((size_t)SELN * 4);
  float*  tem    = (float*)alloc((size_t)TEMN * 4);
  __bf16* temb   = (__bf16*)alloc((size_t)TEMN * 2);
  float*  fmat   = (float*)alloc((size_t)FMATN * 4);
  float*  pmat   = (float*)alloc((size_t)FMATN * 4);
  __bf16* kmat   = (__bf16*)alloc((size_t)KMATP * 2);  // [4608, 2848]
  __bf16* fc1bf  = (__bf16*)alloc((size_t)FC1P * 2);   // [2841, 2848]
  __bf16* hmat   = (__bf16*)alloc((size_t)KMATP * 2);  // [4608, 2848]
  __bf16* fc2bf  = (__bf16*)alloc((size_t)FC2P * 2);   // [2809, 2848]
  __bf16* kpT    = (__bf16*)alloc((size_t)KPTN * 2);   // [6, 2809, 768]
  float*  score  = (float*)alloc((size_t)SCN * 4);

  // 1) BN(x) stats, tconv weights transposed to [N,K], GEMM -> v (relu6+ori fused)
  bn_stats<<<320, 256, 0, stream>>>(x, bnag, bnab, scaleA, shiftA, 2, 784, 784, 320L * 784);
  cvt_tcwT<<<(TCWN + 255) / 256, 256, 0, stream>>>(tconvw, tcwbf);
  GemmP p0 = {}; p0.Af = x; p0.scale = scaleA; p0.shift = shiftA; p0.B = tcwbf;
  p0.bias = tconvb; p0.extra = ori; p0.Cf = vbuf;
  p0.M = 1568; p0.N = 1536; p0.K = 320; p0.ldb = 320;
  gemm_wmma<0><<<dim3(24, 25, 1), 128, 0, stream>>>(p0);

  // 2) BN(v) stats, proj_w is already [N,K], implicit-GEMM conv -> xp
  bn_stats<<<3, 256, 0, stream>>>(vbuf, bnpg, bnpb, scaleP, shiftP, 2, 401408, 401408, 3L * 401408);
  cvt_bf16<<<(PJWN + 255) / 256, 256, 0, stream>>>(projw, pjwbf, PJWN);
  GemmP p1 = {}; p1.Af = vbuf; p1.scale = scaleP; p1.shift = shiftP; p1.B = pjwbf;
  p1.bias = projb; p1.Cf = xp;
  p1.M = 22472; p1.N = 768; p1.K = 2304; p1.ldb = 2304;
  gemm_wmma<1><<<dim3(12, 352, 1), 128, 0, stream>>>(p1);

  // 3) depthwise 3x3x3 pos conv
  posconv_k<<<(XPN + 255) / 256, 256, 0, stream>>>(xp, posw, posb, pos);

  // 4) token selector on frame 0
  bn_stats<<<768, 256, 0, stream>>>(xp, bntg, bntb, scaleT, shiftT, 2, 2809, 11236, 768L * 11236);
  f0bn_k<<<(F0N + 255) / 256, 256, 0, stream>>>(xp, scaleT, shiftT, f0bn);
  dwhs_k<<<(F0N + 255) / 256, 256, 0, stream>>>(f0bn, dww, dwb, dsw);
  sel_k<<<(SELN + 255) / 256, 256, 0, stream>>>(dsw, pww, pwb, sel);
  temg_k<<<64, 256, 0, stream>>>(sel, f0bn, pos, tem, temb, out);

  // 5) chain: BN(xp) stats, build f/p/k matrices, convert fc weights (padded pitch)
  bn_stats<<<768, 256, 0, stream>>>(xp, bncg, bncb, scaleC, shiftC, 2, 11236, 11236, 768L * 11236);
  buildf_k<<<(FMATN + 255) / 256, 256, 0, stream>>>(xp, pos, scaleC, shiftC, fmat, pmat, kmat);
  buildq_k<<<(147456 + 255) / 256, 256, 0, stream>>>(tem, kmat);
  cvt_pad<<<(FC1N + 255) / 256, 256, 0, stream>>>(fc1w, fc1bf, 2841, KP, FC1N);
  cvt_pad<<<(FC2N + 255) / 256, 256, 0, stream>>>(fc2w, fc2bf, 2841, KP, FC2N);

  // 6) fc1 (gelu) -> hmat, fc2 -> kpT (transposed out)
  GemmP p2 = {}; p2.A = kmat; p2.lda = KP; p2.B = fc1bf; p2.ldb = KP;
  p2.bias = fc1b; p2.Cb = hmat; p2.M = 4608; p2.N = 2841; p2.K = 2841;
  gemm_wmma<2><<<dim3(45, 72, 1), 128, 0, stream>>>(p2);
  GemmP p3 = {}; p3.A = hmat; p3.lda = KP; p3.B = fc2bf; p3.ldb = KP;
  p3.bias = fc2b; p3.Cb = kpT; p3.M = 4608; p3.N = 2809; p3.K = 2841;
  gemm_wmma<3><<<dim3(44, 72, 1), 128, 0, stream>>>(p3);

  // 7) score = q @ kp (batched over b,t'), argmax routing gather -> out
  GemmP p4 = {}; p4.A = temb; p4.lda = 768; p4.B = kpT; p4.ldb = 768;
  p4.Cf = score; p4.M = 32; p4.N = 2809; p4.K = 768;
  gemm_wmma<4><<<dim3(44, 1, 6), 128, 0, stream>>>(p4);
  fing_k<<<192, 256, 0, stream>>>(score, fmat, pmat, out);
}
